// DRAW_8710193676298
// MI455X (gfx1250) — compile-verified, matrix operand order checked by reading, not run on hardware
//
#include <hip/hip_runtime.h>
#include <cstdint>
#include <cstddef>

#define DEV __device__ __forceinline__

typedef __attribute__((ext_vector_type(16))) __bf16         v16bf;
typedef __attribute__((ext_vector_type(16))) unsigned short v16u;
typedef __attribute__((ext_vector_type(8)))  unsigned short v8u;
typedef __attribute__((ext_vector_type(8)))  float          v8f;

namespace cfg {
constexpr int Hh = 28, Wd = 28, N5 = 5, Tt = 10, Zz = 10, HD = 256;
constexpr int HW  = Hh * Wd;            // 784
constexpr int Bn  = 16384;
constexpr int MR  = 16;                 // batch rows per workgroup tile
constexpr int NWG = Bn / MR;            // 1024 workgroups
constexpr int NTH = 256;                // 8 wave32 waves

// Encoder combined K: [0,50)=r  [50,64)=pad  [64,320)=h_dec  [320,576)=h_enc
constexpr int KE  = 576;
constexpr int KTE = KE / 32;            // 18 K-tiles
// Decoder combined K: [0,10)=z  [10,32)=pad  [32,288)=h_dec
constexpr int KD  = 288;
constexpr int KTD = KD / 32;            // 9 K-tiles
constexpr int NTL = 64;                 // 1024/16 N-tiles
constexpr int THALF = 512;              // halfs per 32x16 B fragment tile
constexpr size_t ENC_HALFS = (size_t)NTL * KTE * THALF;  // 589824
constexpr size_t DEC_HALFS = (size_t)NTL * KTD * THALF;  // 294912

// ---- LDS layout (float offsets) ----
constexpr int F_SX   = 0;                       // x tile           16*784
constexpr int F_SE   = F_SX   + MR * HW;        // x_err tile       16*784
constexpr int F_HENC = F_SE   + MR * HW;        // 16*256
constexpr int F_CENC = F_HENC + MR * HD;
constexpr int F_HDEC = F_CENC + MR * HD;
constexpr int F_CDEC = F_HDEC + MR * HD;
constexpr int F_GATE = F_CDEC + MR * HD;        // gates            16*1024
constexpr int F_FX   = F_GATE + MR * 1024;      // Fx               16*5*28
constexpr int F_FY   = F_FX   + MR * N5 * 28;   // Fy
constexpr int F_G2   = F_FY   + MR * N5 * 28;   // einsum stage buf
constexpr int F_RB   = F_G2   + MR * N5 * 28;   // r (padded 64)    16*64
constexpr int F_AT   = F_RB   + MR * 64;        // gx,gy,delta,s2,gamma  16*8
constexpr int F_SCR  = F_AT   + MR * 8;         // attn-lin scratch 160
constexpr int F_Z    = F_SCR  + MR * Zz;        // z                160
constexpr int F_KLT  = F_Z    + MR * Zz;        // kl terms         160
constexpr int F_KL   = F_KLT  + MR * Zz;        // kl accum         16
constexpr int F_PT   = F_KL   + MR;             // patch            16*25
constexpr int F_RED  = F_PT   + MR * 25;        // reduction        256
constexpr int F_TOT  = F_RED  + NTH;
constexpr size_t U_SA = (size_t)F_TOT * 4;      // bf16 A staging (16B aligned)
constexpr size_t SMEM_BYTES = U_SA + (size_t)MR * KE * 2;   // ~286 KB < 320 KB/WGP
static_assert((U_SA & 15) == 0, "A staging must be 16B aligned");
}

DEV float sigf(float x) { return 1.f / (1.f + __expf(-x)); }

DEV unsigned short f32_bf16(float f) {
  unsigned int u = __float_as_uint(f);
  u += 0x7fffu + ((u >> 16) & 1u);           // round to nearest even
  return (unsigned short)(u >> 16);
}

DEV v16bf as_bf16(v16u u) { union { v16u u; v16bf b; } c; c.u = u; return c.b; }

// ---------------------------------------------------------------------------
// GEMM core: gates[16 x 1024] += A[16 x (KT*32)] * B, bf16 WMMA, f32 accum.
// A staged in LDS (row-major, 16-bit A layout assembled per lane),
// B pre-packed in global memory in per-fragment lane order.
// ---------------------------------------------------------------------------
DEV void wmma_gates(const unsigned short* __restrict__ Wp,
                    const unsigned short* sAp, float* sgates,
                    int KT, int AK, int tid) {
  const int wave = tid >> 5;
  const int lane = tid & 31;
  const int m    = lane & 15;     // A row / C column
  const int hi   = lane >> 4;     // K-half select / C row-half select
  for (int nt8 = 0; nt8 < 8; ++nt8) {
    const int nt = wave * 8 + nt8;                       // N tile 0..63
    v8f acc = {0.f, 0.f, 0.f, 0.f, 0.f, 0.f, 0.f, 0.f};
    const unsigned short* wp  = Wp + (size_t)nt * KT * 512 + (size_t)lane * 16;
    const unsigned short* ap0 = sAp + m * AK + hi * 8;
    for (int kt = 0; kt < KT; ++kt) {
      v8u alo = *(const v8u*)(ap0 + kt * 32);            // K (i%8)+hi*8
      v8u ahi = *(const v8u*)(ap0 + kt * 32 + 16);       // K 16+(i%8)+hi*8
      v16u au;
#pragma unroll
      for (int i = 0; i < 8; ++i) { au[i] = alo[i]; au[i + 8] = ahi[i]; }
      v16u bu = *(const v16u*)(wp + (size_t)kt * 512);   // 2x b128 global loads
      acc = __builtin_amdgcn_wmma_f32_16x16x32_bf16(
              false, as_bf16(au), false, as_bf16(bu),
              (short)0, acc, false, false);
    }
#pragma unroll
    for (int r8 = 0; r8 < 8; ++r8)
      sgates[(r8 + hi * 8) * 1024 + nt * 16 + m] = acc[r8];
  }
}

// LSTM nonlinearity: gate order i,f,g,o along N (each 256 wide)
DEV void lstm_nl(const float* sgates, const float* __restrict__ bi,
                 const float* __restrict__ bh, float* hS, float* cS, int tid) {
  for (int task = tid; task < cfg::MR * cfg::HD; task += cfg::NTH) {
    int r = task >> 8, n = task & 255;
    const float* g = sgates + r * 1024;
    float gi = g[n]       + bi[n]       + bh[n];
    float gf = g[256 + n] + bi[256 + n] + bh[256 + n];
    float gg = g[512 + n] + bi[512 + n] + bh[512 + n];
    float go = g[768 + n] + bi[768 + n] + bh[768 + n];
    float cn = sigf(gf) * cS[task] + sigf(gi) * tanhf(gg);
    cS[task] = cn;
    hS[task] = sigf(go) * tanhf(cn);
  }
}

// attn linear (h_dec @ Wa.T + ba) -> filterbank params -> normalized Fx/Fy
DEV void attn_and_filter(const float* shdec, const float* __restrict__ Wa,
                         const float* __restrict__ ba, float* scr, float* sat,
                         float* sfx, float* sfy, int tid) {
  using namespace cfg;
  for (int task = tid; task < MR * 5; task += NTH) {
    int r = task / 5, j = task % 5;
    const float* h = shdec + r * HD;
    const float* w = Wa + j * HD;
    float s = ba[j];
    for (int k = 0; k < HD; ++k) s += h[k] * w[k];
    scr[task] = s;
  }
  __syncthreads();
  for (int r = tid; r < MR; r += NTH) {
    float gxt = scr[r*5+0], gyt = scr[r*5+1], lss = scr[r*5+2];
    float ld  = scr[r*5+3], lg  = scr[r*5+4];
    sat[r*8+0] = 0.5f * (float)(Wd + 1) * (tanhf(gxt) + 1.f);
    sat[r*8+1] = 0.5f * (float)(Hh + 1) * (tanhf(gyt) + 1.f);
    sat[r*8+2] = 6.75f * __expf(ld);          // (max(H,W)-1)/(N-1) = 27/4
    sat[r*8+3] = __expf(lss);                 // sigma^2
    sat[r*8+4] = __expf(lg);                  // gamma
  }
  __syncthreads();
  for (int task = tid; task < MR * N5; task += NTH) {
    int r = task / N5, n = task % N5;
    float gx = sat[r*8+0], gy = sat[r*8+1], dl = sat[r*8+2], s2 = sat[r*8+3];
    float mux = gx + (float)(n - 2) * dl;     // rng+1 = n-2
    float muy = gy + (float)(n - 2) * dl;
    float inv2 = 0.5f / s2;
    float* fx = sfx + r * 140 + n * 28;
    float* fy = sfy + r * 140 + n * 28;
    float sxs = 0.f, sys = 0.f;
#pragma unroll
    for (int a = 0; a < 28; ++a) {
      float dx = (float)a - mux, dy = (float)a - muy;
      float vx = __expf(-dx * dx * inv2), vy = __expf(-dy * dy * inv2);
      fx[a] = vx; fy[a] = vy; sxs += vx; sys += vy;
    }
    float ix = 1.f / (sxs + 1e-8f), iy = 1.f / (sys + 1e-8f);
#pragma unroll
    for (int a = 0; a < 28; ++a) { fx[a] *= ix; fy[a] *= iy; }
  }
  __syncthreads();
}

// glimpse: gl[n][m] = gamma * sum_h sum_w Fy[n][h] img[h][w] Fx[m][w]
DEV void read_glimpse(const float* img, const float* sfy, const float* sfx,
                      const float* sat, float* sg2, float* srb, int outoff, int tid) {
  using namespace cfg;
  for (int task = tid; task < MR * 5 * 28; task += NTH) {
    int r = task / 140, rem = task % 140, n = rem / 28, w = rem % 28;
    const float* fy = sfy + r * 140 + n * 28;
    const float* im = img + r * HW + w;
    float s = 0.f;
#pragma unroll
    for (int h = 0; h < 28; ++h) s += fy[h] * im[h * 28];
    sg2[task] = s;
  }
  __syncthreads();
  for (int task = tid; task < MR * 25; task += NTH) {
    int r = task / 25, rem = task % 25, n = rem / 5, mm = rem % 5;
    const float* g  = sg2 + r * 140 + n * 28;
    const float* fx = sfx + r * 140 + mm * 28;
    float s = 0.f;
#pragma unroll
    for (int w = 0; w < 28; ++w) s += g[w] * fx[w];
    srb[r * 64 + outoff + rem] = sat[r*8+4] * s;
  }
  __syncthreads();
}

// ---------------------------------------------------------------------------
__global__ void zero_canvas(float* __restrict__ p, size_t n) {
  size_t i  = (size_t)blockIdx.x * blockDim.x + threadIdx.x;
  size_t st = (size_t)gridDim.x * blockDim.x;
  for (; i < n; i += st) p[i] = 0.f;
}

// Pack LSTM weights fp32 -> bf16 in WMMA B-fragment order.
__global__ void pack_weights(const float* __restrict__ Wihe, const float* __restrict__ Whhe,
                             const float* __restrict__ Wihd, const float* __restrict__ Whhd,
                             unsigned short* __restrict__ enc, unsigned short* __restrict__ dec) {
  using namespace cfg;
  const size_t total_e = ENC_HALFS, total_d = DEC_HALFS;
  size_t idx = (size_t)blockIdx.x * blockDim.x + threadIdx.x;
  size_t st  = (size_t)gridDim.x * blockDim.x;
  for (size_t t = idx; t < total_e + total_d; t += st) {
    bool  is_e = t < total_e;
    size_t v   = is_e ? t : t - total_e;
    int   KT   = is_e ? KTE : KTD;
    int tile   = (int)(v >> 9);
    int lane   = (int)((v >> 4) & 31);
    int i      = (int)(v & 15);
    int nt = tile / KT, kt = tile % KT;
    int n  = nt * 16 + (lane & 15);
    int hi = lane >> 4;
    int k  = kt * 32 + (i & 7) + ((i >> 3) << 4) + hi * 8;
    float w = 0.f;
    if (is_e) {
      if (k < 50)                 w = Wihe[(size_t)n * 306 + k];          // r
      else if (k >= 64 && k < 320) w = Wihe[(size_t)n * 306 + 50 + (k - 64)]; // h_dec
      else if (k >= 320)          w = Whhe[(size_t)n * 256 + (k - 320)];  // h_enc
    } else {
      if (k < 10)                 w = Wihd[(size_t)n * 10 + k];           // z
      else if (k >= 32)           w = Whhd[(size_t)n * 256 + (k - 32)];   // h_dec
    }
    (is_e ? enc : dec)[v] = f32_bf16(w);
  }
}

// ---------------------------------------------------------------------------
__global__ __launch_bounds__(cfg::NTH, 1)
void draw_fused(const float* __restrict__ x, const float* __restrict__ eps,
                const float* __restrict__ Wra, const float* __restrict__ bra,
                const float* __restrict__ Wwa, const float* __restrict__ bwa,
                const float* __restrict__ Wwp, const float* __restrict__ bwp,
                const float* __restrict__ bihe, const float* __restrict__ bhhe,
                const float* __restrict__ bihd, const float* __restrict__ bhhd,
                const float* __restrict__ Wmu, const float* __restrict__ bmu,
                const float* __restrict__ Wlv, const float* __restrict__ blv,
                const unsigned short* __restrict__ encw,
                const unsigned short* __restrict__ decw,
                float* __restrict__ out,          // canvas state + final c
                float* __restrict__ pbce, float* __restrict__ pkl) {
  using namespace cfg;
  extern __shared__ char smem_raw[];
  float* sm    = (float*)smem_raw;
  float* sx    = sm + F_SX;
  float* se    = sm + F_SE;
  float* shenc = sm + F_HENC;
  float* scenc = sm + F_CENC;
  float* shdec = sm + F_HDEC;
  float* scdec = sm + F_CDEC;
  float* sgate = sm + F_GATE;
  float* sfx   = sm + F_FX;
  float* sfy   = sm + F_FY;
  float* sg2   = sm + F_G2;
  float* srb   = sm + F_RB;
  float* sat   = sm + F_AT;
  float* sscr  = sm + F_SCR;
  float* sz    = sm + F_Z;
  float* sklt  = sm + F_KLT;
  float* skl   = sm + F_KL;
  float* spt   = sm + F_PT;
  float* sred  = sm + F_RED;
  unsigned short* sAp = (unsigned short*)(smem_raw + U_SA);

  const int tid  = threadIdx.x;
  const int wg   = blockIdx.x;
  const int row0 = wg * MR;

  // ---- preamble: load x tile, zero states ----
  for (int idx = tid; idx < MR * HW; idx += NTH) {
    int r = idx / HW, p = idx % HW;
    sx[idx] = x[(size_t)(row0 + r) * HW + p];
  }
  for (int idx = tid; idx < MR * HD; idx += NTH) {
    shenc[idx] = 0.f; scenc[idx] = 0.f; shdec[idx] = 0.f; scdec[idx] = 0.f;
  }
  for (int r = tid; r < MR; r += NTH) skl[r] = 0.f;
  __syncthreads();

#pragma unroll 1
  for (int t = 0; t < Tt; ++t) {
    // ---- A: x_err = x - sigmoid(c) ----
    for (int idx = tid; idx < MR * HW; idx += NTH) {
      int r = idx / HW, p = idx % HW;
      float cc = out[(size_t)(row0 + r) * HW + p];
      se[idx] = sx[idx] - sigf(cc);
    }
    __syncthreads();

    // ---- B: read attention filterbank ----
    attn_and_filter(shdec, Wra, bra, sscr, sat, sfx, sfy, tid);

    // ---- C: glimpses of x and x_err -> r (padded to 64) ----
    read_glimpse(sx, sfy, sfx, sat, sg2, srb, 0, tid);
    read_glimpse(se, sfy, sfx, sat, sg2, srb, 25, tid);
    for (int task = tid; task < MR * 14; task += NTH) {
      int r = task / 14; srb[r * 64 + 50 + task % 14] = 0.f;
    }
    __syncthreads();

    // ---- D/E: encoder LSTM via WMMA ----
    for (int task = tid; task < MR * KE; task += NTH) {
      int r = task / KE, k = task % KE;
      float v;
      if (k < 64)       v = srb[r * 64 + k];
      else if (k < 320) v = shdec[r * HD + (k - 64)];
      else              v = shenc[r * HD + (k - 320)];
      sAp[task] = f32_bf16(v);
    }
    __syncthreads();
    wmma_gates(encw, sAp, sgate, KTE, KE, tid);
    __syncthreads();
    lstm_nl(sgate, bihe, bhhe, shenc, scenc, tid);
    __syncthreads();

    // ---- G: mu / logvar / z / KL ----
    for (int task = tid; task < MR * Zz; task += NTH) {
      int r = task / Zz, j = task % Zz;
      const float* h  = shenc + r * HD;
      const float* wm = Wmu + j * HD;
      const float* wl = Wlv + j * HD;
      float mu = bmu[j], lv = blv[j];
      for (int k = 0; k < HD; ++k) { mu += h[k] * wm[k]; lv += h[k] * wl[k]; }
      float e = eps[(size_t)t * Bn * Zz + (size_t)(row0 + r) * Zz + j];
      sz[task]   = mu + __expf(0.5f * lv) * e;
      sklt[task] = -0.5f * (1.f + lv - mu * mu - __expf(lv));
    }
    __syncthreads();
    for (int r = tid; r < MR; r += NTH) {
      float s = 0.f;
      for (int j = 0; j < Zz; ++j) s += sklt[r * Zz + j];
      skl[r] += s;
    }
    __syncthreads();

    // ---- H: decoder LSTM via WMMA ----
    for (int task = tid; task < MR * KD; task += NTH) {
      int r = task / KD, k = task % KD;
      float v;
      if (k < Zz)      v = sz[r * Zz + k];
      else if (k < 32) v = 0.f;
      else             v = shdec[r * HD + (k - 32)];
      sAp[task] = f32_bf16(v);
    }
    __syncthreads();
    wmma_gates(decw, sAp, sgate, KTD, KD, tid);
    __syncthreads();
    lstm_nl(sgate, bihd, bhhd, shdec, scdec, tid);
    __syncthreads();

    // ---- I: write attention -> canvas update ----
    attn_and_filter(shdec, Wwa, bwa, sscr, sat, sfx, sfy, tid);
    for (int task = tid; task < MR * 25; task += NTH) {
      int r = task / 25, pidx = task % 25;
      const float* h = shdec + r * HD;
      const float* w = Wwp + pidx * HD;
      float s = bwp[pidx];
      for (int k = 0; k < HD; ++k) s += h[k] * w[k];
      spt[task] = s;
    }
    __syncthreads();
    // stage1: P2[n][w] = sum_m patch[n][m] * Fx[m][w]
    for (int task = tid; task < MR * 5 * 28; task += NTH) {
      int r = task / 140, rem = task % 140, n = rem / 28, w = rem % 28;
      float s = 0.f;
#pragma unroll
      for (int mm = 0; mm < 5; ++mm)
        s += spt[r * 25 + n * 5 + mm] * sfx[r * 140 + mm * 28 + w];
      sg2[task] = s;
    }
    __syncthreads();
    // stage2: c[h][w] += (1/(gamma+eps)) * sum_n Fy[n][h] * P2[n][w]
    for (int task = tid; task < MR * HW; task += NTH) {
      int r = task / HW, p = task % HW, hh = p / 28, w = p % 28;
      float invg = 1.f / (sat[r*8+4] + 1e-8f);
      float s = 0.f;
#pragma unroll
      for (int n = 0; n < 5; ++n)
        s += sfy[r * 140 + n * 28 + hh] * sg2[r * 140 + n * 28 + w];
      out[(size_t)(row0 + r) * HW + p] += invg * s;
    }
    __threadfence();
    __syncthreads();
  }

  // ---- BCE + per-WG partials (deterministic) ----
  float wg_bce = 0.f;
  for (int r = 0; r < MR; ++r) {
    float local = 0.f;
    for (int p = tid; p < HW; p += NTH) {
      float cc = out[(size_t)(row0 + r) * HW + p];
      float xx = sx[r * HW + p];
      local += fmaxf(cc, 0.f) - cc * xx + log1pf(__expf(-fabsf(cc)));
    }
    sred[tid] = local;
    __syncthreads();
    for (int s = NTH / 2; s > 0; s >>= 1) {
      if (tid < s) sred[tid] += sred[tid + s];
      __syncthreads();
    }
    if (tid == 0) wg_bce += sred[0];
    __syncthreads();
  }
  if (tid == 0) {
    float wg_kl = 0.f;
    for (int r = 0; r < MR; ++r) wg_kl += skl[r];
    pbce[wg] = wg_bce;
    pkl[wg]  = wg_kl;
  }
}

__global__ void finalize(const float* __restrict__ pb, const float* __restrict__ pk,
                         float* __restrict__ out3) {
  using namespace cfg;
  __shared__ float sb[256], sk[256];
  float b = 0.f, k = 0.f;
  for (int i = threadIdx.x; i < NWG; i += 256) { b += pb[i]; k += pk[i]; }
  sb[threadIdx.x] = b; sk[threadIdx.x] = k;
  __syncthreads();
  for (int s = 128; s > 0; s >>= 1) {
    if ((int)threadIdx.x < s) { sb[threadIdx.x] += sb[threadIdx.x + s]; sk[threadIdx.x] += sk[threadIdx.x + s]; }
    __syncthreads();
  }
  if (threadIdx.x == 0) {
    float bm = sb[0] / (float)Bn, km = sk[0] / (float)Bn;
    out3[0] = bm + km;   // nelbo.mean()
    out3[1] = bm;        // bce.mean()
    out3[2] = km;        // kl.mean()
  }
}

// ---------------------------------------------------------------------------
extern "C" void kernel_launch(void* const* d_in, const int* in_sizes, int n_in,
                              void* d_out, int out_size, void* d_ws, size_t ws_size,
                              hipStream_t stream) {
  using namespace cfg;
  (void)in_sizes; (void)n_in; (void)out_size; (void)ws_size;

  const float* x    = (const float*)d_in[0];
  const float* eps  = (const float*)d_in[1];
  const float* Wra  = (const float*)d_in[2];
  const float* bra  = (const float*)d_in[3];
  const float* Wwa  = (const float*)d_in[4];
  const float* bwa  = (const float*)d_in[5];
  const float* Wwp  = (const float*)d_in[6];
  const float* bwp  = (const float*)d_in[7];
  const float* Wihe = (const float*)d_in[8];
  const float* Whhe = (const float*)d_in[9];
  const float* bihe = (const float*)d_in[10];
  const float* bhhe = (const float*)d_in[11];
  const float* Wihd = (const float*)d_in[12];
  const float* Whhd = (const float*)d_in[13];
  const float* bihd = (const float*)d_in[14];
  const float* bhhd = (const float*)d_in[15];
  const float* Wmu  = (const float*)d_in[16];
  const float* bmu  = (const float*)d_in[17];
  const float* Wlv  = (const float*)d_in[18];
  const float* blv  = (const float*)d_in[19];

  unsigned short* encw = (unsigned short*)d_ws;
  unsigned short* decw = encw + ENC_HALFS;
  float* pbce = (float*)((char*)d_ws + (ENC_HALFS + DEC_HALFS) * 2);
  float* pkl  = pbce + NWG;
  float* out  = (float*)d_out;

  (void)hipFuncSetAttribute((const void*)draw_fused,
                            hipFuncAttributeMaxDynamicSharedMemorySize,
                            (int)SMEM_BYTES);

  zero_canvas<<<2048, 256, 0, stream>>>(out, (size_t)Bn * HW + 3);
  pack_weights<<<1024, 256, 0, stream>>>(Wihe, Whhe, Wihd, Whhd, encw, decw);
  draw_fused<<<NWG, NTH, SMEM_BYTES, stream>>>(
      x, eps, Wra, bra, Wwa, bwa, Wwp, bwp,
      bihe, bhhe, bihd, bhhd, Wmu, bmu, Wlv, blv,
      encw, decw, out, pbce, pkl);
  finalize<<<1, 256, 0, stream>>>(pbce, pkl, out + (size_t)Bn * HW);
}
